// HierarchicalTransformerClassifier_9852654977796
// MI455X (gfx1250) — matrix operand order repaired and checked by reading, not compiled, not run on hardware
//
#include <hip/hip_runtime.h>
#include <math.h>

// ---------------------------------------------------------------------------
// HierarchicalTransformerClassifier for MI455X (gfx1250, wave32, WMMA bf16)
//
// Sizes: B=32, N=4096, M=256, K=16, D=128, DFF=256, H=4 (hd=32), NCLS=40
//
// Input index map (setup_inputs() dict insertion order, flattened recursively):
//   0           x                       (32*3*4096)
//   1..4        embed: W1(3x64) b1 W2(64x128) b2
//   5..20       local layer   (16 tensors, order below)
//   21..36      global[0]
//   37..52      global[1]
//   53..68      global[2]
//   69..82      cls: W1 b1 bn1_g bn1_b bn1_rm bn1_rv W2 b2 bn2_g bn2_b bn2_rm bn2_rv W3 b3
// Layer tensor order: Wq bq Wk bk Wv bv Wo bo ln1_g ln1_b ffn_W1 ffn_b1 ffn_W2 ffn_b2 ln2_g ln2_b
// ---------------------------------------------------------------------------

typedef __attribute__((ext_vector_type(16))) __bf16 v16bf;
typedef __attribute__((ext_vector_type(8)))  __bf16 v8bf;
typedef __attribute__((ext_vector_type(8)))  float  v8f;

union BfTile { v16bf v; v8bf h[2]; __bf16 e[16]; };

__device__ __forceinline__ v8f wmma_bf16(v8f acc, v16bf a, v16bf b) {
  // D = A(16x32) * B(32x16) + C, f32 accumulate
  return __builtin_amdgcn_wmma_f32_16x16x32_bf16(
      /*neg_a*/false, a, /*neg_b*/false, b, /*c_mod*/(short)0, acc,
      /*reuse_a*/false, /*reuse_b*/false);
}

// A-tile 16(M)x32(K), row-major bf16 source, per CDNA5 16-bit A layout:
// lanes 0-15: M=lane, K chunks {0..7} and {16..23}; lanes 16-31: {8..15},{24..31}
__device__ __forceinline__ v16bf wmma_load_a(const __bf16* src, int lda) {
  int lane = threadIdx.x & 31;
  int row  = lane & 15;
  int kh   = (lane & 16) ? 8 : 0;
  BfTile t;
  t.h[0] = *(const v8bf*)(src + row * lda + kh);
  t.h[1] = *(const v8bf*)(src + row * lda + 16 + kh);
  return t.v;
}

// B-tile 32(K)x16(N) loaded from PRE-TRANSPOSED weights Bt (N x K row-major):
// lanes 0-15: N=lane, K 0..7,16..23; lanes 16-31: N=lane-16, K 8..15,24..31
__device__ __forceinline__ v16bf wmma_load_bt(const __bf16* bt, int ldb) {
  int lane = threadIdx.x & 31;
  int n  = lane & 15;
  int k0 = (lane & 16) ? 8 : 0;
  BfTile t;
  t.h[0] = *(const v8bf*)(bt + n * ldb + k0);
  t.h[1] = *(const v8bf*)(bt + n * ldb + k0 + 16);
  return t.v;
}

// B-tile from a row-major K x N source (e.g. V matrix), strided reads,
// zero-filled beyond kvalid (for seq-16 local attention padding).
__device__ __forceinline__ v16bf wmma_load_b_strided(const __bf16* b, int ldbn, int kvalid) {
  int lane = threadIdx.x & 31;
  int n  = lane & 15;
  int k0 = (lane & 16) ? 8 : 0;
  BfTile t;
#pragma unroll
  for (int i = 0; i < 16; ++i) {
    int k = k0 + (i & 7) + ((i & 8) ? 16 : 0);
    t.e[i] = (k < kvalid) ? b[k * ldbn + n] : (__bf16)0.0f;
  }
  return t.v;
}

// A-tile converted on the fly from f32 (LDS score rows), zero beyond kvalid
__device__ __forceinline__ v16bf wmma_load_a_f32(const float* src, int lda, int kvalid) {
  int lane = threadIdx.x & 31;
  int row  = lane & 15;
  int kh   = (lane & 16) ? 8 : 0;
  BfTile t;
#pragma unroll
  for (int j = 0; j < 8; ++j) {
    int kA = kh + j, kB = 16 + kh + j;
    t.e[j]     = (kA < kvalid) ? (__bf16)src[row * lda + kA] : (__bf16)0.0f;
    t.e[8 + j] = (kB < kvalid) ? (__bf16)src[row * lda + kB] : (__bf16)0.0f;
  }
  return t.v;
}

// ---------------------------------------------------------------------------
// Weight convert: W (Kd x N, f32, row-major) -> Wt (N x Kd, bf16, row-major)
// ---------------------------------------------------------------------------
__global__ void conv_wt_kernel(const float* __restrict__ W, __bf16* __restrict__ Wt,
                               int Kd, int N) {
  int i = blockIdx.x * 256 + threadIdx.x;
  if (i < N * Kd) {
    int n = i / Kd, k = i % Kd;
    Wt[i] = (__bf16)W[k * N + n];
  }
}

// ---------------------------------------------------------------------------
// Farthest point sampling: one block per batch, min-dist lives in registers
// ---------------------------------------------------------------------------
__global__ void fps_kernel(const float* __restrict__ x, int* __restrict__ cidx) {
  int b = blockIdx.x, t = threadIdx.x;
  __shared__ float bv[256];
  __shared__ int   bi[256];
  __shared__ int   curs;
  const float* xb = x + b * 3 * 4096;
  float md[16];
#pragma unroll
  for (int i = 0; i < 16; ++i) md[i] = 1e10f;
  int cur = 0;
  if (t == 0) cidx[b * 256] = 0;
  for (int m = 1; m < 256; ++m) {
    float cx = xb[cur], cy = xb[4096 + cur], cz = xb[8192 + cur];
    float bestv = -1.0f; int besti = 0;
#pragma unroll
    for (int i = 0; i < 16; ++i) {
      int n = t + 256 * i;
      float dx = xb[n] - cx, dy = xb[4096 + n] - cy, dz = xb[8192 + n] - cz;
      float d = dx * dx + dy * dy + dz * dz;
      md[i] = fminf(md[i], d);
      if (md[i] > bestv) { bestv = md[i]; besti = n; }
    }
    bv[t] = bestv; bi[t] = besti;
    __syncthreads();
    if (t == 0) {
      float mv = bv[0]; int mi = bi[0];
      for (int k = 1; k < 256; ++k) if (bv[k] > mv) { mv = bv[k]; mi = bi[k]; }
      curs = mi;
      cidx[b * 256 + m] = mi;
    }
    __syncthreads();
    cur = curs;
  }
}

// ---------------------------------------------------------------------------
// kNN (K=16 nearest of 4096): one block per (b,m)
// ---------------------------------------------------------------------------
__global__ void knn_kernel(const float* __restrict__ x, const int* __restrict__ cidx,
                           int* __restrict__ nn) {
  int bm = blockIdx.x;
  int b = bm >> 8;
  int t = threadIdx.x;
  __shared__ float dist[4096];
  __shared__ float bv[256];
  __shared__ int   bi[256];
  const float* xb = x + b * 3 * 4096;
  int ci = cidx[bm];
  float cx = xb[ci], cy = xb[4096 + ci], cz = xb[8192 + ci];
#pragma unroll
  for (int i = 0; i < 16; ++i) {
    int n = t + 256 * i;
    float dx = xb[n] - cx, dy = xb[4096 + n] - cy, dz = xb[8192 + n] - cz;
    dist[n] = dx * dx + dy * dy + dz * dz;
  }
  __syncthreads();
  for (int k = 0; k < 16; ++k) {
    float bvv = 1e30f; int bii = 0;
#pragma unroll
    for (int i = 0; i < 16; ++i) {
      int n = t + 256 * i;
      float d = dist[n];
      if (d < bvv) { bvv = d; bii = n; }
    }
    bv[t] = bvv; bi[t] = bii;
    __syncthreads();
    if (t == 0) {
      float mv = bv[0]; int mi = bi[0];
      for (int j = 1; j < 256; ++j) if (bv[j] < mv) { mv = bv[j]; mi = bi[j]; }
      nn[bm * 16 + k] = mi;
      dist[mi] = 1e30f;
    }
    __syncthreads();
  }
}

// ---------------------------------------------------------------------------
// Fused local stage: gather + embed MLP + encoder layer (seq 16) + maxpool
// One 256-thread block (8 waves) per (b,m); everything WMMA in LDS.
// ---------------------------------------------------------------------------
struct LocalParams {
  const float* x; const int* cidx; const int* nn;
  const float* eW1; const float* eb1; const __bf16* eW2t; const float* eb2;
  const __bf16* Wqt; const __bf16* Wkt; const __bf16* Wvt; const __bf16* Wot;
  const float* bq; const float* bk; const float* bvv; const float* bo;
  const float* ln1g; const float* ln1b;
  const __bf16* W1t; const float* fb1; const __bf16* W2t; const float* fb2;
  const float* ln2g; const float* ln2b;
  float* outF; __bf16* outB;
};

struct __align__(16) LocalSmem {
  float  hf[16][128];     // f32 master activations
  float  t2f[16][128];    // pre-LN temp (also stashes gathered coords)
  float  sc4[4][16][16];  // per-head scores
  float  mu[16], inv[16];
  __bf16 h1b[16][64];
  __bf16 hb[16][128];
  __bf16 qb[16][128];
  __bf16 kb[16][128];
  __bf16 vb[16][128];
  __bf16 pb[4][16][32];   // softmax probs, zero-padded to K=32
  __bf16 ab[16][128];     // attention output
  __bf16 tb[16][256];     // FFN mid
};

__global__ void local_fused_kernel(LocalParams p) {
  __shared__ LocalSmem sm;
  int bm = blockIdx.x;
  int b = bm >> 8;
  int t = threadIdx.x, wave = t >> 5, lane = t & 31;
  const float* xb = p.x + b * 3 * 4096;
  int ci = p.cidx[bm];

  // 1) gather + center (stash coords in t2f row 0)
  if (t < 48) {
    int kpt = t / 3, c = t % 3;
    int idx = p.nn[bm * 16 + kpt];
    sm.t2f[0][t] = xb[c * 4096 + idx] - xb[c * 4096 + ci];
  }
  __syncthreads();

  // 2) embed layer 1 (16x3 @ 3x64, relu) -> h1b
  {
    int o0 = t * 4;
#pragma unroll
    for (int u = 0; u < 4; ++u) {
      int o = o0 + u, r = o >> 6, n = o & 63;
      float s = p.eb1[n];
#pragma unroll
      for (int c = 0; c < 3; ++c) s += sm.t2f[0][r * 3 + c] * p.eW1[c * 64 + n];
      sm.h1b[r][n] = (__bf16)fmaxf(s, 0.0f);
    }
  }
  __syncthreads();

  // 3) embed layer 2 (16x64 @ 64x128) via WMMA, wave w -> n-tile w
  {
    int n0 = wave * 16;
    v8f acc = {};
    for (int k = 0; k < 64; k += 32) {
      v16bf a  = wmma_load_a(&sm.h1b[0][k], 64);
      v16bf bt = wmma_load_bt(p.eW2t + n0 * 64 + k, 64);
      acc = wmma_bf16(acc, a, bt);
    }
    int col = lane & 15, r0 = (lane & 16) ? 8 : 0;
    float bias = p.eb2[n0 + col];
#pragma unroll
    for (int r = 0; r < 8; ++r) {
      float v = acc[r] + bias;
      sm.hf[r0 + r][n0 + col] = v;
      sm.hb[r0 + r][n0 + col] = (__bf16)v;
    }
  }
  __syncthreads();

  // 4) Q,K,V projections (q pre-scaled by 1/sqrt(hd))
  {
    const __bf16* wts[3] = { p.Wqt, p.Wkt, p.Wvt };
    const float*  bss[3] = { p.bq,  p.bk,  p.bvv };
    __bf16 (*outs[3])[128] = { sm.qb, sm.kb, sm.vb };
    int n0 = wave * 16;
    int col = lane & 15, r0 = (lane & 16) ? 8 : 0;
    for (int mat = 0; mat < 3; ++mat) {
      v8f acc = {};
      for (int k = 0; k < 128; k += 32) {
        v16bf a  = wmma_load_a(&sm.hb[0][k], 128);
        v16bf bt = wmma_load_bt(wts[mat] + n0 * 128 + k, 128);
        acc = wmma_bf16(acc, a, bt);
      }
      float bias = bss[mat][n0 + col];
      float scale = (mat == 0) ? 0.17677669529663687f : 1.0f;
#pragma unroll
      for (int r = 0; r < 8; ++r)
        outs[mat][r0 + r][n0 + col] = (__bf16)((acc[r] + bias) * scale);
    }
  }
  __syncthreads();

  // 5) scores per head (16x16, K=32): waves 0..3
  if (wave < 4) {
    int h = wave;
    v16bf a  = wmma_load_a(&sm.qb[0][h * 32], 128);
    v16bf bt = wmma_load_bt(&sm.kb[0][h * 32], 128);
    v8f s = {};
    s = wmma_bf16(s, a, bt);
    int col = lane & 15, r0 = (lane & 16) ? 8 : 0;
#pragma unroll
    for (int r = 0; r < 8; ++r) sm.sc4[h][r0 + r][col] = s[r];
  }
  __syncthreads();

  // 6) softmax over 16 keys; write probs zero-padded to width 32
  if (t < 64) {
    int h = t >> 4, r = t & 15;
    float mx = -1e30f;
#pragma unroll
    for (int c = 0; c < 16; ++c) mx = fmaxf(mx, sm.sc4[h][r][c]);
    float e[16]; float sum = 0.0f;
#pragma unroll
    for (int c = 0; c < 16; ++c) { e[c] = __expf(sm.sc4[h][r][c] - mx); sum += e[c]; }
    float inv = 1.0f / sum;
#pragma unroll
    for (int c = 0; c < 16; ++c) sm.pb[h][r][c] = (__bf16)(e[c] * inv);
#pragma unroll
    for (int c = 16; c < 32; ++c) sm.pb[h][r][c] = (__bf16)0.0f;
  }
  __syncthreads();

  // 7) attn x V per head: waves 0..3, padded K=32 single WMMA per 16-col half
  if (wave < 4) {
    int h = wave;
    int col = lane & 15, r0 = (lane & 16) ? 8 : 0;
    for (int nh = 0; nh < 2; ++nh) {
      v16bf a  = wmma_load_a(&sm.pb[h][0][0], 32);
      v16bf bv = wmma_load_b_strided(&sm.vb[0][h * 32 + nh * 16], 128, 16);
      v8f acc = {};
      acc = wmma_bf16(acc, a, bv);
#pragma unroll
      for (int r = 0; r < 8; ++r)
        sm.ab[r0 + r][h * 32 + nh * 16 + col] = (__bf16)acc[r];
    }
  }
  __syncthreads();

  // 8) O-projection + residual -> t2f
  {
    int n0 = wave * 16;
    v8f acc = {};
    for (int k = 0; k < 128; k += 32) {
      v16bf a  = wmma_load_a(&sm.ab[0][k], 128);
      v16bf bt = wmma_load_bt(p.Wot + n0 * 128 + k, 128);
      acc = wmma_bf16(acc, a, bt);
    }
    int col = lane & 15, r0 = (lane & 16) ? 8 : 0;
    float bias = p.bo[n0 + col];
#pragma unroll
    for (int r = 0; r < 8; ++r)
      sm.t2f[r0 + r][n0 + col] = sm.hf[r0 + r][n0 + col] + acc[r] + bias;
  }
  __syncthreads();

  // 9) LayerNorm 1 -> hf, hb
  if (t < 16) {
    float s = 0.0f;
    for (int c = 0; c < 128; ++c) s += sm.t2f[t][c];
    float mu = s * (1.0f / 128.0f), v = 0.0f;
    for (int c = 0; c < 128; ++c) { float d = sm.t2f[t][c] - mu; v += d * d; }
    sm.mu[t] = mu; sm.inv[t] = rsqrtf(v * (1.0f / 128.0f) + 1e-5f);
  }
  __syncthreads();
  for (int i = t; i < 2048; i += 256) {
    int r = i >> 7, c = i & 127;
    float y = (sm.t2f[r][c] - sm.mu[r]) * sm.inv[r] * p.ln1g[c] + p.ln1b[c];
    sm.hf[r][c] = y; sm.hb[r][c] = (__bf16)y;
  }
  __syncthreads();

  // 10) FFN1 (16x128 @ 128x256, relu) -> tb
  for (int it = 0; it < 2; ++it) {
    int n0 = (wave + it * 8) * 16;
    v8f acc = {};
    for (int k = 0; k < 128; k += 32) {
      v16bf a  = wmma_load_a(&sm.hb[0][k], 128);
      v16bf bt = wmma_load_bt(p.W1t + n0 * 128 + k, 128);
      acc = wmma_bf16(acc, a, bt);
    }
    int col = lane & 15, r0 = (lane & 16) ? 8 : 0;
    float bias = p.fb1[n0 + col];
#pragma unroll
    for (int r = 0; r < 8; ++r)
      sm.tb[r0 + r][n0 + col] = (__bf16)fmaxf(acc[r] + bias, 0.0f);
  }
  __syncthreads();

  // 11) FFN2 (16x256 @ 256x128) + residual -> t2f
  {
    int n0 = wave * 16;
    v8f acc = {};
    for (int k = 0; k < 256; k += 32) {
      v16bf a  = wmma_load_a(&sm.tb[0][k], 256);
      v16bf bt = wmma_load_bt(p.W2t + n0 * 256 + k, 256);
      acc = wmma_bf16(acc, a, bt);
    }
    int col = lane & 15, r0 = (lane & 16) ? 8 : 0;
    float bias = p.fb2[n0 + col];
#pragma unroll
    for (int r = 0; r < 8; ++r)
      sm.t2f[r0 + r][n0 + col] = sm.hf[r0 + r][n0 + col] + acc[r] + bias;
  }
  __syncthreads();

  // 12) LayerNorm 2 -> hf
  if (t < 16) {
    float s = 0.0f;
    for (int c = 0; c < 128; ++c) s += sm.t2f[t][c];
    float mu = s * (1.0f / 128.0f), v = 0.0f;
    for (int c = 0; c < 128; ++c) { float d = sm.t2f[t][c] - mu; v += d * d; }
    sm.mu[t] = mu; sm.inv[t] = rsqrtf(v * (1.0f / 128.0f) + 1e-5f);
  }
  __syncthreads();
  for (int i = t; i < 2048; i += 256) {
    int r = i >> 7, c = i & 127;
    sm.hf[r][c] = (sm.t2f[r][c] - sm.mu[r]) * sm.inv[r] * p.ln2g[c] + p.ln2b[c];
  }
  __syncthreads();

  // 13) maxpool over 16 points -> (f32 + bf16) local feature
  if (t < 128) {
    float mx = sm.hf[0][t];
#pragma unroll
    for (int r = 1; r < 16; ++r) mx = fmaxf(mx, sm.hf[r][t]);
    p.outF[bm * 128 + t] = mx;
    p.outB[bm * 128 + t] = (__bf16)mx;
  }
}

// ---------------------------------------------------------------------------
// Generic WMMA GEMM with async-LDS weight staging.
// Block = 256 threads (8 waves) computes a 128(row) x 64(col) output panel.
// The 64xKd weight panel is staged once into LDS via the CDNA5 async path
// (global_load_async_to_lds_b128 + s_wait_asynccnt) and shared by all waves.
// Each wave keeps 4 independent accumulators (16x64) so WMMA chains interleave.
// ---------------------------------------------------------------------------
__global__ void gemm_bf16_kernel(const __bf16* __restrict__ A, const __bf16* __restrict__ Bt,
                                 const float* __restrict__ bias,
                                 float* outF, __bf16* outB,
                                 int Kd, int N, int relu, float scale) {
  __shared__ __attribute__((aligned(16))) __bf16 bt_s[64 * 256];  // up to Kd=256
  int wave = threadIdx.x >> 5, lane = threadIdx.x & 31;
  int ncg = N >> 6;                         // 64-col groups
  int br  = blockIdx.x / ncg;               // row block (128 rows)
  int cg  = blockIdx.x - br * ncg;          // col group

  // --- async stage of the 64 x Kd bf16 weight panel into LDS ---
  {
    const __bf16* gpanel = Bt + (size_t)cg * 64 * Kd;
    int nchunk = (64 * Kd) >> 3;            // 8 bf16 (16B) per chunk; exact /256
    for (int c = threadIdx.x; c < nchunk; c += 256) {
      unsigned long long g = (unsigned long long)(gpanel + c * 8);
      unsigned l = (unsigned)(unsigned long long)(bt_s + c * 8);
      asm volatile("global_load_async_to_lds_b128 %0, %1, off"
                   :: "v"(l), "v"(g) : "memory");
    }
    asm volatile("s_wait_asynccnt 0x0" ::: "memory");
  }
  __syncthreads();

  int tm = br * 8 + wave;                   // 16-row tile index for this wave
  const __bf16* arow = A + (size_t)tm * 16 * Kd;
  v8f acc0 = {}, acc1 = {}, acc2 = {}, acc3 = {};
  for (int k = 0; k < Kd; k += 32) {
    __builtin_prefetch(arow + k + 64, 0, 1);   // global_prefetch_b8
    v16bf a  = wmma_load_a(arow + k, Kd);
    v16bf b0 = wmma_load_bt(bt_s + 0  * Kd + k, Kd);
    v16bf b1 = wmma_load_bt(bt_s + 16 * Kd + k, Kd);
    v16bf b2 = wmma_load_bt(bt_s + 32 * Kd + k, Kd);
    v16bf b3 = wmma_load_bt(bt_s + 48 * Kd + k, Kd);
    acc0 = wmma_bf16(acc0, a, b0);
    acc1 = wmma_bf16(acc1, a, b1);
    acc2 = wmma_bf16(acc2, a, b2);
    acc3 = wmma_bf16(acc3, a, b3);
  }
  int col = lane & 15, r0 = (lane & 16) ? 8 : 0;
  v8f accs[4] = { acc0, acc1, acc2, acc3 };
#pragma unroll
  for (int s = 0; s < 4; ++s) {
    int nb = cg * 64 + s * 16;
    float bi = bias[nb + col];
#pragma unroll
    for (int r = 0; r < 8; ++r) {
      float v = (accs[s][r] + bi) * scale;
      if (relu) v = fmaxf(v, 0.0f);
      size_t idx = (size_t)(tm * 16 + r0 + r) * N + nb + col;
      if (outF) outF[idx] = v;
      if (outB) outB[idx] = (__bf16)v;
    }
  }
}

// ---------------------------------------------------------------------------
// Global attention: blockDim=128 (4 waves), each wave owns one (b,h,q-tile).
// Scores via 16 WMMAs (K=32), softmax in per-wave LDS, PxV via 16 WMMAs.
// ---------------------------------------------------------------------------
__global__ void attn_global_kernel(const __bf16* __restrict__ Qb, const __bf16* __restrict__ Kb,
                                   const __bf16* __restrict__ Vb, __bf16* __restrict__ Ob) {
  __shared__ float sc[4][16][256];   // 64 KB
  int wave = threadIdx.x >> 5, lane = threadIdx.x & 31;
  int qt = blockIdx.x * 4 + wave;    // 0..2047
  int b   = qt >> 6;                 // 64 tiles per batch (4 heads * 16 q-tiles)
  int rem = qt & 63;
  int h   = rem >> 4;
  int q0  = (rem & 15) * 16;
  int col = lane & 15, r0 = (lane & 16) ? 8 : 0;

  v16bf aq = wmma_load_a(Qb + (size_t)(b * 256 + q0) * 128 + h * 32, 128);

  for (int j = 0; j < 16; ++j) {
    v16bf bk = wmma_load_bt(Kb + (size_t)(b * 256 + j * 16) * 128 + h * 32, 128);
    v8f s = {};
    s = wmma_bf16(s, aq, bk);
#pragma unroll
    for (int r = 0; r < 8; ++r) sc[wave][r0 + r][j * 16 + col] = s[r];
  }

  // softmax (q already scaled by 1/sqrt(hd) in the Q GEMM)
  for (int r = 0; r < 16; ++r) {
    float* row = sc[wave][r];
    float mx = -1e30f;
    for (int i = lane; i < 256; i += 32) mx = fmaxf(mx, row[i]);
    for (int m = 16; m; m >>= 1) mx = fmaxf(mx, __shfl_xor(mx, m, 32));
    float sum = 0.0f;
    for (int i = lane; i < 256; i += 32) { float e = __expf(row[i] - mx); row[i] = e; sum += e; }
    for (int m = 16; m; m >>= 1) sum += __shfl_xor(sum, m, 32);
    float inv = 1.0f / sum;
    for (int i = lane; i < 256; i += 32) row[i] *= inv;
  }

  // out = P(16x256) @ V(256x32), two 16-col halves
  for (int nh = 0; nh < 2; ++nh) {
    v8f acc = {};
    for (int j = 0; j < 8; ++j) {
      v16bf ap = wmma_load_a_f32(&sc[wave][0][j * 32], 256, 32);
      v16bf bv = wmma_load_b_strided(Vb + (size_t)(b * 256 + j * 32) * 128 + h * 32 + nh * 16,
                                     128, 32);
      acc = wmma_bf16(acc, ap, bv);
    }
#pragma unroll
    for (int r = 0; r < 8; ++r)
      Ob[(size_t)(b * 256 + q0 + r0 + r) * 128 + h * 32 + nh * 16 + col] = (__bf16)acc[r];
  }
}

// ---------------------------------------------------------------------------
// Residual + LayerNorm: one wave32 per row of (R,128); writes f32 + bf16
// ---------------------------------------------------------------------------
__global__ void ln_res_kernel(float* __restrict__ xf, const float* __restrict__ tf,
                              const float* __restrict__ g, const float* __restrict__ be,
                              __bf16* __restrict__ xb) {
  int row  = blockIdx.x * 8 + (threadIdx.x >> 5);
  int lane = threadIdx.x & 31;
  float v[4]; float s = 0.0f;
#pragma unroll
  for (int i = 0; i < 4; ++i) {
    int c = lane + 32 * i;
    v[i] = xf[(size_t)row * 128 + c] + tf[(size_t)row * 128 + c];
    s += v[i];
  }
  for (int m = 16; m; m >>= 1) s += __shfl_xor(s, m, 32);
  float mu = s * 0.0078125f;
  float var = 0.0f;
#pragma unroll
  for (int i = 0; i < 4; ++i) { float d = v[i] - mu; var += d * d; }
  for (int m = 16; m; m >>= 1) var += __shfl_xor(var, m, 32);
  float inv = rsqrtf(var * 0.0078125f + 1e-5f);
#pragma unroll
  for (int i = 0; i < 4; ++i) {
    int c = lane + 32 * i;
    float y = (v[i] - mu) * inv * g[c] + be[c];
    xf[(size_t)row * 128 + c] = y;
    xb[(size_t)row * 128 + c] = (__bf16)y;
  }
}

// ---------------------------------------------------------------------------
// Max over sequence (32,256,128) -> (32,128)
// ---------------------------------------------------------------------------
__global__ void gmax_kernel(const float* __restrict__ xf, float* __restrict__ gmax) {
  int i = blockIdx.x * 256 + threadIdx.x;   // 0..4095
  int b = i >> 7, c = i & 127;
  float mx = -1e30f;
  for (int s = 0; s < 256; ++s) mx = fmaxf(mx, xf[(size_t)(b * 256 + s) * 128 + c]);
  gmax[i] = mx;
}

// ---------------------------------------------------------------------------
// Classifier head: MLP + eval-mode BN + log_softmax (single block)
// ---------------------------------------------------------------------------
struct ClsParams {
  const float* gmax;
  const float* W1; const float* b1;
  const float* g1; const float* be1; const float* rm1; const float* rv1;
  const float* W2; const float* b2;
  const float* g2; const float* be2; const float* rm2; const float* rv2;
  const float* W3; const float* b3;
  float* out;
};

__global__ void cls_kernel(ClsParams p) {
  __shared__ float z1[32][256];
  __shared__ float z2[32][128];
  __shared__ float lg[32][40];
  int t = threadIdx.x;
  for (int i = t; i < 32 * 256; i += 256) {
    int r = i >> 8, n = i & 255;
    float s = p.b1[n];
    for (int k = 0; k < 128; ++k) s += p.gmax[r * 128 + k] * p.W1[k * 256 + n];
    s = (s - p.rm1[n]) * rsqrtf(p.rv1[n] + 1e-5f) * p.g1[n] + p.be1[n];
    z1[r][n] = fmaxf(s, 0.0f);
  }
  __syncthreads();
  for (int i = t; i < 32 * 128; i += 256) {
    int r = i >> 7, n = i & 127;
    float s = p.b2[n];
    for (int k = 0; k < 256; ++k) s += z1[r][k] * p.W2[k * 128 + n];
    s = (s - p.rm2[n]) * rsqrtf(p.rv2[n] + 1e-5f) * p.g2[n] + p.be2[n];
    z2[r][n] = fmaxf(s, 0.0f);
  }
  __syncthreads();
  for (int i = t; i < 32 * 40; i += 256) {
    int r = i / 40, n = i % 40;
    float s = p.b3[n];
    for (int k = 0; k < 128; ++k) s += z2[r][k] * p.W3[k * 40 + n];
    lg[r][n] = s;
  }
  __syncthreads();
  if (t < 32) {
    float mx = -1e30f;
    for (int n = 0; n < 40; ++n) mx = fmaxf(mx, lg[t][n]);
    float sum = 0.0f;
    for (int n = 0; n < 40; ++n) sum += __expf(lg[t][n] - mx);
    float lse = mx + __logf(sum);
    for (int n = 0; n < 40; ++n) p.out[t * 40 + n] = lg[t][n] - lse;
  }
}

// ---------------------------------------------------------------------------
// Host orchestration
// ---------------------------------------------------------------------------
enum { IN_X = 0, IN_EW1 = 1, IN_EB1 = 2, IN_EW2 = 3, IN_EB2 = 4,
       IN_LOCAL = 5, IN_G0 = 21, IN_G1 = 37, IN_G2 = 53, IN_CLS = 69 };
enum { L_WQ = 0, L_BQ, L_WK, L_BK, L_WV, L_BV, L_WO, L_BO, L_LN1G, L_LN1B,
       L_W1, L_B1, L_W2, L_B2, L_LN2G, L_LN2B };

extern "C" void kernel_launch(void* const* d_in, const int* in_sizes, int n_in,
                              void* d_out, int out_size, void* d_ws, size_t ws_size,
                              hipStream_t stream) {
  (void)in_sizes; (void)n_in; (void)out_size; (void)ws_size;
  auto F = [&](int i) { return (const float*)d_in[i]; };

  char* wp = (char*)d_ws;
  auto alloc = [&](size_t bytes) -> void* {
    void* r = (void*)wp; wp += (bytes + 255) & ~(size_t)255; return r;
  };

  // bf16 transposed weight copies
  __bf16* eW2t = (__bf16*)alloc(128 * 64 * 2);
  struct LayerW { __bf16 *wqt, *wkt, *wvt, *wot, *w1t, *w2t; } lw[4];
  const int layerBase[4] = { IN_LOCAL, IN_G0, IN_G1, IN_G2 };
  for (int l = 0; l < 4; ++l) {
    lw[l].wqt = (__bf16*)alloc(128 * 128 * 2);
    lw[l].wkt = (__bf16*)alloc(128 * 128 * 2);
    lw[l].wvt = (__bf16*)alloc(128 * 128 * 2);
    lw[l].wot = (__bf16*)alloc(128 * 128 * 2);
    lw[l].w1t = (__bf16*)alloc(256 * 128 * 2);   // (DFF x D)
    lw[l].w2t = (__bf16*)alloc(128 * 256 * 2);   // (D x DFF)
  }
  int*    cidx = (int*)alloc(32 * 256 * 4);
  int*    nn   = (int*)alloc(32 * 256 * 16 * 4);
  float*  xf   = (float*)alloc((size_t)8192 * 128 * 4);
  __bf16* xb   = (__bf16*)alloc((size_t)8192 * 128 * 2);
  __bf16* qb   = (__bf16*)alloc((size_t)8192 * 128 * 2);
  __bf16* kb   = (__bf16*)alloc((size_t)8192 * 128 * 2);
  __bf16* vb   = (__bf16*)alloc((size_t)8192 * 128 * 2);
  __bf16* ab   = (__bf16*)alloc((size_t)8192 * 128 * 2);
  float*  tf   = (float*)alloc((size_t)8192 * 128 * 4);
  __bf16* f1b  = (__bf16*)alloc((size_t)8192 * 256 * 2);
  float*  gmax = (float*)alloc(32 * 128 * 4);

  // --- weight conversion (transpose + f32->bf16) ---
  conv_wt_kernel<<<(128 * 64 + 255) / 256, 256, 0, stream>>>(F(IN_EW2), eW2t, 64, 128);
  for (int l = 0; l < 4; ++l) {
    int base = layerBase[l];
    conv_wt_kernel<<<64, 256, 0, stream>>>(F(base + L_WQ), lw[l].wqt, 128, 128);
    conv_wt_kernel<<<64, 256, 0, stream>>>(F(base + L_WK), lw[l].wkt, 128, 128);
    conv_wt_kernel<<<64, 256, 0, stream>>>(F(base + L_WV), lw[l].wvt, 128, 128);
    conv_wt_kernel<<<64, 256, 0, stream>>>(F(base + L_WO), lw[l].wot, 128, 128);
    conv_wt_kernel<<<128, 256, 0, stream>>>(F(base + L_W1), lw[l].w1t, 128, 256);
    conv_wt_kernel<<<128, 256, 0, stream>>>(F(base + L_W2), lw[l].w2t, 256, 128);
  }

  // --- sampling / grouping ---
  fps_kernel<<<32, 256, 0, stream>>>(F(IN_X), cidx);
  knn_kernel<<<32 * 256, 256, 0, stream>>>(F(IN_X), cidx, nn);

  // --- fused local stage -> (B*M, 128) features ---
  {
    int base = IN_LOCAL;
    LocalParams lp;
    lp.x = F(IN_X); lp.cidx = cidx; lp.nn = nn;
    lp.eW1 = F(IN_EW1); lp.eb1 = F(IN_EB1); lp.eW2t = eW2t; lp.eb2 = F(IN_EB2);
    lp.Wqt = lw[0].wqt; lp.Wkt = lw[0].wkt; lp.Wvt = lw[0].wvt; lp.Wot = lw[0].wot;
    lp.bq = F(base + L_BQ); lp.bk = F(base + L_BK); lp.bvv = F(base + L_BV); lp.bo = F(base + L_BO);
    lp.ln1g = F(base + L_LN1G); lp.ln1b = F(base + L_LN1B);
    lp.W1t = lw[0].w1t; lp.fb1 = F(base + L_B1);
    lp.W2t = lw[0].w2t; lp.fb2 = F(base + L_B2);
    lp.ln2g = F(base + L_LN2G); lp.ln2b = F(base + L_LN2B);
    lp.outF = xf; lp.outB = xb;
    local_fused_kernel<<<32 * 256, 256, 0, stream>>>(lp);
  }

  // --- 3 global encoder layers over (32, 256, 128) ---
  // GEMM grids: 64 row-blocks x (N/64) col groups
  const float qscale = 0.17677669529663687f;  // 1/sqrt(32)
  for (int l = 1; l < 4; ++l) {
    int base = layerBase[l];
    gemm_bf16_kernel<<<128, 256, 0, stream>>>(xb, lw[l].wqt, F(base + L_BQ),
                                              nullptr, qb, 128, 128, 0, qscale);
    gemm_bf16_kernel<<<128, 256, 0, stream>>>(xb, lw[l].wkt, F(base + L_BK),
                                              nullptr, kb, 128, 128, 0, 1.0f);
    gemm_bf16_kernel<<<128, 256, 0, stream>>>(xb, lw[l].wvt, F(base + L_BV),
                                              nullptr, vb, 128, 128, 0, 1.0f);
    attn_global_kernel<<<512, 128, 0, stream>>>(qb, kb, vb, ab);
    gemm_bf16_kernel<<<128, 256, 0, stream>>>(ab, lw[l].wot, F(base + L_BO),
                                              tf, nullptr, 128, 128, 0, 1.0f);
    ln_res_kernel<<<1024, 256, 0, stream>>>(xf, tf, F(base + L_LN1G), F(base + L_LN1B), xb);
    gemm_bf16_kernel<<<256, 256, 0, stream>>>(xb, lw[l].w1t, F(base + L_B1),
                                              nullptr, f1b, 128, 256, 1, 1.0f);
    gemm_bf16_kernel<<<128, 256, 0, stream>>>(f1b, lw[l].w2t, F(base + L_B2),
                                              tf, nullptr, 256, 128, 0, 1.0f);
    ln_res_kernel<<<1024, 256, 0, stream>>>(xf, tf, F(base + L_LN2G), F(base + L_LN2B), xb);
  }

  // --- global maxpool + classifier ---
  gmax_kernel<<<16, 256, 0, stream>>>(xf, gmax);
  {
    int c = IN_CLS;
    ClsParams cp;
    cp.gmax = gmax;
    cp.W1 = F(c + 0);  cp.b1 = F(c + 1);
    cp.g1 = F(c + 2);  cp.be1 = F(c + 3); cp.rm1 = F(c + 4); cp.rv1 = F(c + 5);
    cp.W2 = F(c + 6);  cp.b2 = F(c + 7);
    cp.g2 = F(c + 8);  cp.be2 = F(c + 9); cp.rm2 = F(c + 10); cp.rv2 = F(c + 11);
    cp.W3 = F(c + 12); cp.b3 = F(c + 13);
    cp.out = (float*)d_out;
    cls_kernel<<<1, 256, 0, stream>>>(cp);
  }
}
